// CasualSelfAttention_90159953477863
// MI455X (gfx1250) — compile-verified, matrix operand order checked by reading, not run on hardware
//
#include <hip/hip_runtime.h>
#include <hip/hip_bf16.h>

// ---- problem constants (match reference) ----
#define T_SEQ   2048
#define C_DIM   1024
#define NH      16
#define HD      64
#define BATCH   2
#define M_ROWS  (BATCH * T_SEQ)     // 4096

typedef __attribute__((ext_vector_type(16))) __bf16 v16bf;
typedef __attribute__((ext_vector_type(8)))  __bf16 v8bf;
typedef __attribute__((ext_vector_type(4)))  __bf16 v4bf;
typedef __attribute__((ext_vector_type(8)))  float  v8f;

typedef unsigned int u32x4 __attribute__((ext_vector_type(4)));
typedef int          i32x4 __attribute__((ext_vector_type(4)));
typedef int          i32x8 __attribute__((ext_vector_type(8)));

union Frag { v16bf v; v8bf h[2]; };

#define LOG2E 1.4426950408889634f

#if __has_builtin(__builtin_amdgcn_tensor_load_to_lds)
#define USE_TDM 1
#else
#define USE_TDM 0
#endif

// ---------------------------------------------------------------------------
// fp32 -> bf16 cast (vectorized x4)
// ---------------------------------------------------------------------------
__global__ __launch_bounds__(256)
void cast_f32_to_bf16(const float* __restrict__ src, __bf16* __restrict__ dst, int n4) {
    int i = blockIdx.x * blockDim.x + threadIdx.x;
    if (i < n4) {
        float4 f = ((const float4*)src)[i];
        v4bf o;
        o[0] = (__bf16)f.x; o[1] = (__bf16)f.y; o[2] = (__bf16)f.z; o[3] = (__bf16)f.w;
        ((v4bf*)dst)[i] = o;
    }
}

#if USE_TDM
// Issue one TDM 2-D tile load: 128 rows x 32 bf16 from A[row0..row0+127][kb..kb+31]
// into LDS at lds_off, with LDS padding 4 DWORDs every 16 DWORDs (row stride 40
// elements, matching the WMMA A-fragment layout used by the compute loop).
__device__ static inline void tdm_load_tile(const __bf16* gsrc, unsigned lds_off) {
    unsigned long long ga = (unsigned long long)(const void*)gsrc;
    u32x4 g0;
    g0[0] = 1u;                                                   // count=1, user mode
    g0[1] = lds_off;                                              // lds_addr (bytes)
    g0[2] = (unsigned)ga;                                         // global_addr[31:0]
    g0[3] = (unsigned)((ga >> 32) & 0x1FFFFFFull) | (2u << 30);   // addr[56:32] | type=2
    i32x8 g1;
    g1[0] = (1 << 16)        // data_size = 2 bytes
          | (1 << 20)        // pad_enable
          | (3 << 22)        // pad_interval: 16 DWORDs (one 64B tile row)
          | (3 << 25);       // pad_amount: 4 DWORDs (8 bf16) -> row stride 40 elems
    g1[1] = (int)((C_DIM & 0xFFFF) << 16);                        // tensor_dim0 lo16
    g1[2] = (int)((C_DIM >> 16) | ((M_ROWS & 0xFFFF) << 16));     // tdim0 hi | tdim1 lo
    g1[3] = (int)((M_ROWS >> 16) | (32 << 16));                   // tdim1 hi | tile_dim0=32
    g1[4] = 128;                                                  // tile_dim1=128, tile_dim2=0
    g1[5] = C_DIM;                                                // tensor_dim0_stride lo32
    g1[6] = 0;
    g1[7] = 0;
    i32x4 gz = {0, 0, 0, 0};
#if defined(__clang_major__) && (__clang_major__ >= 23)
    i32x8 gz8 = {0, 0, 0, 0, 0, 0, 0, 0};
    __builtin_amdgcn_tensor_load_to_lds(g0, g1, gz, gz, gz8, 0);
#else
    __builtin_amdgcn_tensor_load_to_lds(g0, g1, gz, gz, 0);
#endif
}

__device__ static inline void tdm_wait0() {
#if __has_builtin(__builtin_amdgcn_s_wait_tensorcnt)
    __builtin_amdgcn_s_wait_tensorcnt(0);
#else
    asm volatile("s_wait_tensorcnt 0x0" ::: "memory");
#endif
}
#endif // USE_TDM

// ---------------------------------------------------------------------------
// Tiled WMMA GEMM:  out[M,N] = A[M,K] @ W[N,K]^T    (K = C_DIM = 1024)
// block = 256 threads (8 waves), macro tile 128(M) x 128(N),
// wave tile 64(M) x 32(N) -> 8 x v_wmma_f32_16x16x32_bf16 per K-step.
// A tile staged in LDS by the Tensor Data Mover (double-buffered, DMA of
// tile k+1 overlaps WMMA on tile k). W fragments load straight from global
// (row-major over K => contiguous 16B B-fragment chunks).
// mode 0: write bf16 into [B,H,T,D] (QKV projection)
// mode 1: write fp32 [M,C] + bias   (output projection)
// ---------------------------------------------------------------------------
__global__ __launch_bounds__(256)
void gemm_bf16_wt(const __bf16* __restrict__ A,
                  const __bf16* __restrict__ W,
                  __bf16* __restrict__ out_qkv,
                  float*  __restrict__ out_f32,
                  const float* __restrict__ bias,
                  int mode)
{
    __shared__ __align__(16) __bf16 XT[2][128 * 40];  // padded stride 40 (80B)

    const int tid  = threadIdx.x;
    const int w    = tid >> 5;
    const int lane = tid & 31;
    const int lrow = lane & 15;
    const int half = lane >> 4;
    const int h8   = half * 8;

    const int mbase = blockIdx.y * 128;
    const int nbase = blockIdx.x * 128;
    const int wm = (w >> 2) * 64;       // 0 / 64
    const int wn = (w & 3) * 32;        // 0 / 32 / 64 / 96

    const v8f vz = {0,0,0,0,0,0,0,0};
    v8f acc[4][2];
    #pragma unroll
    for (int i = 0; i < 4; i++)
        #pragma unroll
        for (int j = 0; j < 2; j++) acc[i][j] = vz;

    const int NITER = C_DIM / 32;

#if USE_TDM
    const unsigned xt_off[2] = { (unsigned)(size_t)(&XT[0][0]),
                                 (unsigned)(size_t)(&XT[1][0]) };
    if (w == 0) {
        tdm_load_tile(A + (size_t)mbase * C_DIM, xt_off[0]);
    }
#endif

    for (int it = 0; it < NITER; ++it) {
        const int kb  = it * 32;
        const int cur = it & 1;

#if USE_TDM
        if (w == 0) tdm_wait0();     // buffer `cur` fully written by the TDM
        __syncthreads();             // publish buf[cur]; buf[cur^1] reads finished
        if (w == 0 && (it + 1) < NITER) {
            tdm_load_tile(A + (size_t)mbase * C_DIM + (kb + 32), xt_off[cur ^ 1]);
        }
#else
        __syncthreads();
        // cooperative A-tile load fallback: 128 rows x 32 k, 512 chunks of 8 bf16
        #pragma unroll
        for (int i = 0; i < 2; i++) {
            int c   = i * 256 + tid;
            int row = c >> 2;
            int d0  = (c & 3) * 8;
            *(v8bf*)(&XT[cur][row * 40 + d0]) =
                *(const v8bf*)(A + (size_t)(mbase + row) * C_DIM + kb + d0);
        }
        __syncthreads();
#endif

        // B fragments (W rows): per-lane col n fixed, K chunks contiguous in global
        Frag bfr[2];
        #pragma unroll
        for (int ns = 0; ns < 2; ns++) {
            const __bf16* wp = W + (size_t)(nbase + wn + ns * 16 + lrow) * C_DIM + kb;
            bfr[ns].h[0] = *(const v8bf*)(wp + h8);
            bfr[ns].h[1] = *(const v8bf*)(wp + 16 + h8);
            if (kb + 32 < C_DIM) __builtin_prefetch(wp + 32, 0, 1);
        }

        #pragma unroll
        for (int ms = 0; ms < 4; ms++) {
            Frag afr;
            const __bf16* xp = &XT[cur][(wm + ms * 16 + lrow) * 40];
            afr.h[0] = *(const v8bf*)(xp + h8);
            afr.h[1] = *(const v8bf*)(xp + 16 + h8);
            #pragma unroll
            for (int ns = 0; ns < 2; ns++)
                acc[ms][ns] = __builtin_amdgcn_wmma_f32_16x16x32_bf16(
                    false, afr.v, false, bfr[ns].v, (short)0, acc[ms][ns], false, false);
        }
    }

    // epilogue: C/D layout = lane holds column n = lane%16, rows m = half*8 + e
    #pragma unroll
    for (int ms = 0; ms < 4; ms++) {
        #pragma unroll
        for (int ns = 0; ns < 2; ns++) {
            int n = nbase + wn + ns * 16 + lrow;
            #pragma unroll
            for (int e = 0; e < 8; e++) {
                int   m  = mbase + wm + ms * 16 + h8 + e;
                float vv = acc[ms][ns][e];
                if (mode == 0) {
                    int b = m >> 11, t = m & (T_SEQ - 1);
                    int hh = n >> 6, d = n & (HD - 1);
                    out_qkv[((size_t)(b * NH + hh) * T_SEQ + t) * HD + d] = (__bf16)vv;
                } else {
                    out_f32[(size_t)m * C_DIM + n] = vv + bias[n];
                }
            }
        }
    }
}

// ---------------------------------------------------------------------------
// Flash attention, causal, scale = 1/sqrt(C) = 1/32 exactly.
// grid = (T/64 qblocks, B*H), block = 128 (4 waves, 16 q-rows each).
// Q/K fragments straight from global (row-major over D => contiguous chunks);
// V transposed into LDS for the PV B-operand; P re-shaped via per-wave LDS.
// ---------------------------------------------------------------------------
__global__ __launch_bounds__(128)
void attn_flash(const __bf16* __restrict__ Q, const __bf16* __restrict__ K,
                const __bf16* __restrict__ V, __bf16* __restrict__ att)
{
    __shared__ __align__(16) __bf16 VT[64 * 72];     // V^T tile: [d][k], stride 72
    __shared__ __align__(16) __bf16 Pw[4][16 * 72];  // per-wave P staging, stride 72

    const int tid  = threadIdx.x;
    const int w    = tid >> 5;
    const int lane = tid & 31;
    const int lrow = lane & 15;
    const int half = lane >> 4;
    const int h8   = half * 8;

    const int bh   = blockIdx.y;                 // 0..31
    const int b    = bh >> 4, h = bh & (NH - 1);
    const int qblk = blockIdx.x;                 // 0..31
    const int qw   = qblk * 64 + w * 16;         // this wave's first q row
    const size_t headbase = (size_t)bh * T_SEQ * HD;

    // Q A-fragments (2 K-tiles over D=64): row = lane%16, chunks at h8 / 16+h8
    Frag qa[2];
    {
        const __bf16* qp = Q + headbase + (size_t)(qw + lrow) * HD;
        #pragma unroll
        for (int kt = 0; kt < 2; kt++) {
            qa[kt].h[0] = *(const v8bf*)(qp + kt * 32 + h8);
            qa[kt].h[1] = *(const v8bf*)(qp + kt * 32 + 16 + h8);
        }
    }

    const v8f vz = {0,0,0,0,0,0,0,0};
    float mi[8], li[8];
    v8f o[4];
    #pragma unroll
    for (int e = 0; e < 8; e++) { mi[e] = -__builtin_inff(); li[e] = 0.f; }
    #pragma unroll
    for (int dt = 0; dt < 4; dt++) o[dt] = vz;

    for (int kb = 0; kb <= qblk; ++kb) {
        __syncthreads();
        // cooperative transpose-load of the 64x64 V tile into VT[d][k]
        #pragma unroll
        for (int i = 0; i < 4; i++) {
            int c   = i * 128 + tid;
            int row = c >> 3;
            int d0  = (c & 7) * 8;
            v8bf vv = *(const v8bf*)(V + headbase + (size_t)(kb * 64 + row) * HD + d0);
            #pragma unroll
            for (int j = 0; j < 8; j++) VT[(d0 + j) * 72 + row] = vv[j];
        }
        __syncthreads();

        // ---- S = Q K^T  (4 n-tiles of 16 keys, K-dim = D = 64 -> 2 wmma each) ----
        v8f s[4];
        #pragma unroll
        for (int nt = 0; nt < 4; nt++) s[nt] = vz;
        #pragma unroll
        for (int nt = 0; nt < 4; nt++) {
            const __bf16* kp = K + headbase + (size_t)(kb * 64 + nt * 16 + lrow) * HD;
            #pragma unroll
            for (int kt = 0; kt < 2; kt++) {
                Frag bf_;
                bf_.h[0] = *(const v8bf*)(kp + kt * 32 + h8);
                bf_.h[1] = *(const v8bf*)(kp + kt * 32 + 16 + h8);
                s[nt] = __builtin_amdgcn_wmma_f32_16x16x32_bf16(
                    false, qa[kt].v, false, bf_.v, (short)0, s[nt], false, false);
            }
        }

        // ---- scale + causal mask (C/D layout: n = lane%16, m = half*8+e) ----
        float sv[4][8];
        #pragma unroll
        for (int nt = 0; nt < 4; nt++) {
            #pragma unroll
            for (int e = 0; e < 8; e++) {
                float x = s[nt][e] * 0.03125f;   // 1/sqrt(1024)
                if (kb == qblk) {
                    int kidx = kb * 64 + nt * 16 + lrow;
                    int qidx = qw + h8 + e;
                    if (kidx > qidx) x = -__builtin_inff();
                }
                sv[nt][e] = x;
            }
        }

        // ---- online softmax: row reductions across the 16 lanes of each half ----
        #pragma unroll
        for (int e = 0; e < 8; e++) {
            float rmax = fmaxf(fmaxf(sv[0][e], sv[1][e]), fmaxf(sv[2][e], sv[3][e]));
            #pragma unroll
            for (int off = 1; off < 16; off <<= 1)
                rmax = fmaxf(rmax, __shfl_xor(rmax, off, 32));
            float mnew  = fmaxf(mi[e], rmax);
            float alpha = __builtin_exp2f((mi[e] - mnew) * LOG2E);
            float rsum  = 0.f;
            #pragma unroll
            for (int nt = 0; nt < 4; nt++) {
                float p = __builtin_exp2f((sv[nt][e] - mnew) * LOG2E);
                sv[nt][e] = p;
                rsum += p;
            }
            #pragma unroll
            for (int off = 1; off < 16; off <<= 1)
                rsum += __shfl_xor(rsum, off, 32);
            li[e] = li[e] * alpha + rsum;
            mi[e] = mnew;
            #pragma unroll
            for (int dt = 0; dt < 4; dt++) o[dt][e] = o[dt][e] * alpha;
        }

        // ---- stage P (C-layout) -> LDS, re-read in A-layout ----
        __bf16* pw = Pw[w];
        #pragma unroll
        for (int nt = 0; nt < 4; nt++)
            #pragma unroll
            for (int e = 0; e < 8; e++)
                pw[(h8 + e) * 72 + nt * 16 + lrow] = (__bf16)sv[nt][e];

        // ---- O += P @ V  (K-dim = 64 keys -> 2 wmma per d-tile) ----
        #pragma unroll
        for (int kt = 0; kt < 2; kt++) {
            Frag pa;
            const __bf16* pp = pw + lrow * 72 + kt * 32;
            pa.h[0] = *(const v8bf*)(pp + h8);
            pa.h[1] = *(const v8bf*)(pp + 16 + h8);
            #pragma unroll
            for (int dt = 0; dt < 4; dt++) {
                Frag vb;
                const __bf16* vp = &VT[(dt * 16 + lrow) * 72 + kt * 32];
                vb.h[0] = *(const v8bf*)(vp + h8);
                vb.h[1] = *(const v8bf*)(vp + 16 + h8);
                o[dt] = __builtin_amdgcn_wmma_f32_16x16x32_bf16(
                    false, pa.v, false, vb.v, (short)0, o[dt], false, false);
            }
        }
    }

    // ---- epilogue: normalize and write att[B,T,C] as bf16 ----
    #pragma unroll
    for (int e = 0; e < 8; e++) {
        float inv = 1.f / li[e];
        int t = qw + h8 + e;
        #pragma unroll
        for (int dt = 0; dt < 4; dt++) {
            int col = h * HD + dt * 16 + lrow;
            att[(size_t)(b * T_SEQ + t) * C_DIM + col] = (__bf16)(o[dt][e] * inv);
        }
    }
}

// ---------------------------------------------------------------------------
// host-side launch
// ---------------------------------------------------------------------------
extern "C" void kernel_launch(void* const* d_in, const int* in_sizes, int n_in,
                              void* d_out, int out_size, void* d_ws, size_t ws_size,
                              hipStream_t stream) {
    const float* x  = (const float*)d_in[0];
    const float* Wq = (const float*)d_in[1];
    const float* Wk = (const float*)d_in[2];
    const float* Wv = (const float*)d_in[3];
    const float* Wo = (const float*)d_in[4];
    const float* bo = (const float*)d_in[5];
    float* out = (float*)d_out;

    // workspace layout (bf16 elements); total ~48 MB
    __bf16* ws    = (__bf16*)d_ws;
    __bf16* x_bf  = ws;                        // 4096*1024
    __bf16* wq_bf = x_bf  + (size_t)M_ROWS * C_DIM;
    __bf16* wk_bf = wq_bf + (size_t)C_DIM * C_DIM;
    __bf16* wv_bf = wk_bf + (size_t)C_DIM * C_DIM;
    __bf16* wo_bf = wv_bf + (size_t)C_DIM * C_DIM;
    __bf16* q_bf  = wo_bf + (size_t)C_DIM * C_DIM;   // [B,H,T,D]
    __bf16* k_bf  = q_bf  + (size_t)M_ROWS * C_DIM;
    __bf16* v_bf  = k_bf  + (size_t)M_ROWS * C_DIM;
    __bf16* a_bf  = v_bf  + (size_t)M_ROWS * C_DIM;  // [B,T,C]

    const int nx = M_ROWS * C_DIM;   // 4194304
    const int nw = C_DIM * C_DIM;    // 1048576
    cast_f32_to_bf16<<<nx / 4 / 256, 256, 0, stream>>>(x,  x_bf,  nx / 4);
    cast_f32_to_bf16<<<nw / 4 / 256, 256, 0, stream>>>(Wq, wq_bf, nw / 4);
    cast_f32_to_bf16<<<nw / 4 / 256, 256, 0, stream>>>(Wk, wk_bf, nw / 4);
    cast_f32_to_bf16<<<nw / 4 / 256, 256, 0, stream>>>(Wv, wv_bf, nw / 4);
    cast_f32_to_bf16<<<nw / 4 / 256, 256, 0, stream>>>(Wo, wo_bf, nw / 4);

    dim3 gg(C_DIM / 128, M_ROWS / 128);        // (8, 32)
    gemm_bf16_wt<<<gg, 256, 0, stream>>>(x_bf, wq_bf, q_bf, nullptr, nullptr, 0);
    gemm_bf16_wt<<<gg, 256, 0, stream>>>(x_bf, wk_bf, k_bf, nullptr, nullptr, 0);
    gemm_bf16_wt<<<gg, 256, 0, stream>>>(x_bf, wv_bf, v_bf, nullptr, nullptr, 0);

    dim3 ga(T_SEQ / 64, BATCH * NH);           // (32, 32)
    attn_flash<<<ga, 128, 0, stream>>>(q_bf, k_bf, v_bf, a_bf);

    gemm_bf16_wt<<<gg, 256, 0, stream>>>(a_bf, wo_bf, nullptr, out, bo, 1);
}
